// MOE_51771535786339
// MI455X (gfx1250) — matrix operand order repaired and checked by reading, not compiled, not run on hardware
//
#include <hip/hip_runtime.h>

#define N_TOK   16384
#define D_MODEL 1024
#define NEXP    8
#define TOPK    2
#define CAP     4096
#define D_FF    3072
#define D_HALF  1536
#define NSLOT   (N_TOK * TOPK)
#define LSTR    72   // padded LDS row stride in halfwords (144B -> conflict-free b128)

typedef __attribute__((ext_vector_type(16))) __bf16 v16bf;
typedef __attribute__((ext_vector_type(8)))  __bf16 v8bf;
typedef __attribute__((ext_vector_type(8)))  float  v8f;
typedef int v4i_vs __attribute__((vector_size(16)));   // matches builtin param type

#define AS1 __attribute__((address_space(1)))
#define AS3 __attribute__((address_space(3)))

#if defined(__has_builtin)
#  if __has_builtin(__builtin_amdgcn_global_load_async_to_lds_b128)
#    define HAVE_ASYNC_B128 1
#  endif
#endif
#ifndef HAVE_ASYNC_B128
#  define HAVE_ASYNC_B128 0
#endif

// Async global->LDS copy of 16 bytes (per lane). base must be wave-uniform in
// the asm path (GVS mode: SGPR base + 32-bit VGPR byte offset).
static __device__ __forceinline__ void async_b128(const unsigned short* base,
                                                  unsigned eoff,          // element offset
                                                  unsigned short* l) {
#if HAVE_ASYNC_B128
  __builtin_amdgcn_global_load_async_to_lds_b128(
      (AS1 v4i_vs*)(base + eoff), (AS3 v4i_vs*)l, 0, 0);
#else
  unsigned loff = (unsigned)(uintptr_t)(AS3 void*)l;
  asm volatile("global_load_async_to_lds_b128 %0, %1, %2"
               :: "v"(loff), "v"(eoff * 2u), "s"(base)
               : "memory");
#endif
}

static __device__ __forceinline__ void async_wait0() {
#if defined(__has_builtin) && __has_builtin(__builtin_amdgcn_s_wait_asynccnt)
  __builtin_amdgcn_s_wait_asynccnt(0);
#else
  asm volatile("s_wait_asynccnt 0x0" ::: "memory");
#endif
}

static __device__ __forceinline__ unsigned short f2bf(float f) {
  union { float f; unsigned u; } v; v.f = f;
  unsigned r = v.u + 0x7FFFu + ((v.u >> 16) & 1u);   // round-to-nearest-even
  return (unsigned short)(r >> 16);
}

// ---- WMMA fragment loads from LDS (bf16, 16x32 A / 32x16 B layouts) ----
// A: lanes 0-15 row M=l, K {0..7} then {16..23}; lanes 16-31 row M=l-16, K {8..15},{24..31}
static __device__ __forceinline__ v16bf load_afrag(const unsigned short* s, int row, int kk) {
  int l = threadIdx.x & 31;
  const unsigned short* p = s + (row + (l & 15)) * LSTR + kk + ((l >> 4) << 3);
  v8bf lo = *(const v8bf*)p;
  v8bf hi = *(const v8bf*)(p + 16);
  return __builtin_shufflevector(lo, hi, 0,1,2,3,4,5,6,7,8,9,10,11,12,13,14,15);
}
// B: lanes 0-15 col N=l, K {0..7} then {16..23}; lanes 16-31 col N=l-16, K {8..15},{24..31}
static __device__ __forceinline__ v16bf load_bfrag(const unsigned short* s, int col, int kk) {
  int l = threadIdx.x & 31;
  const unsigned short* p = s + (col + (l & 15)) * LSTR + kk + ((l >> 4) << 3);
  v8bf lo = *(const v8bf*)p;
  v8bf hi = *(const v8bf*)(p + 16);
  return __builtin_shufflevector(lo, hi, 0,1,2,3,4,5,6,7,8,9,10,11,12,13,14,15);
}
static __device__ __forceinline__ v8f wmma_bf16(v16bf a, v16bf b, v8f c) {
  return __builtin_amdgcn_wmma_f32_16x16x32_bf16(false, a, false, b, (short)0, c, false, false);
}

// ---- weight transpose + f32->bf16 convert: in [e][R][C] -> out [e][C][R] ----
__global__ void __launch_bounds__(256) transpose_convert(const float* __restrict__ in,
                                                         unsigned short* __restrict__ out,
                                                         int R, int C) {
  __shared__ float tile[32][33];
  int e = blockIdx.z;
  const float* ip = in + (size_t)e * R * C;
  unsigned short* op = out + (size_t)e * R * C;
  int c0 = blockIdx.x * 32, r0 = blockIdx.y * 32;
  int tx = threadIdx.x, ty = threadIdx.y;
  #pragma unroll
  for (int i = 0; i < 4; i++)
    tile[ty + i * 8][tx] = ip[(size_t)(r0 + ty + i * 8) * C + c0 + tx];
  __syncthreads();
  #pragma unroll
  for (int i = 0; i < 4; i++)
    op[(size_t)(c0 + ty + i * 8) * R + r0 + tx] = f2bf(tile[tx][ty + i * 8]);
}

// ---- gate: softmax(x @ w_gate) top-2, one wave per token ----
__global__ void __launch_bounds__(256) gate_kernel(const float* __restrict__ x,
                                                   const float* __restrict__ wg,
                                                   int* __restrict__ tk_idx,
                                                   float* __restrict__ tk_w) {
  __shared__ float sW[D_MODEL * NEXP];
  int t = threadIdx.x;
  for (int i = t; i < D_MODEL * NEXP; i += 256) sW[i] = wg[i];
  __syncthreads();
  int wid = t >> 5, lane = t & 31;
  int tok = blockIdx.x * 8 + wid;
  const float* xp = x + (size_t)tok * D_MODEL;
  float acc[NEXP] = {};
  for (int k0 = 0; k0 < D_MODEL; k0 += 32) {
    float xv = xp[k0 + lane];
    const float* wr = sW + (k0 + lane) * NEXP;
    #pragma unroll
    for (int e = 0; e < NEXP; e++) acc[e] += xv * wr[e];
  }
  #pragma unroll
  for (int off = 16; off; off >>= 1)
    #pragma unroll
    for (int e = 0; e < NEXP; e++) acc[e] += __shfl_down(acc[e], off, 32);
  if (lane == 0) {
    float m = acc[0];
    #pragma unroll
    for (int e = 1; e < NEXP; e++) m = fmaxf(m, acc[e]);
    float p[NEXP], s = 0.f;
    #pragma unroll
    for (int e = 0; e < NEXP; e++) { p[e] = __expf(acc[e] - m); s += p[e]; }
    float inv = 1.f / s;
    #pragma unroll
    for (int e = 0; e < NEXP; e++) p[e] *= inv;
    int i0 = 0;
    #pragma unroll
    for (int e = 1; e < NEXP; e++) if (p[e] > p[i0]) i0 = e;
    int i1 = (i0 == 0) ? 1 : 0;
    #pragma unroll
    for (int e = 0; e < NEXP; e++) if (e != i0 && p[e] > p[i1]) i1 = e;
    tk_idx[2 * tok] = i0; tk_idx[2 * tok + 1] = i1;
    tk_w[2 * tok] = p[i0]; tk_w[2 * tok + 1] = p[i1];
  }
}

// ---- routing positions: block histogram -> scan -> ordered rank ----
__global__ void hist_kernel(const int* __restrict__ tk_idx, int* __restrict__ blkcnt) {
  __shared__ int c[NEXP];
  int t = threadIdx.x;
  if (t < NEXP) c[t] = 0;
  __syncthreads();
  int base = blockIdx.x * 1024;
  for (int j = 0; j < 4; j++) atomicAdd(&c[tk_idx[base + j * 256 + t]], 1);
  __syncthreads();
  if (t < NEXP) blkcnt[blockIdx.x * NEXP + t] = c[t];
}

__global__ void scan_kernel(const int* __restrict__ blkcnt, int* __restrict__ blkbase) {
  int e = threadIdx.x;
  if (e < NEXP) {
    int b = 0;
    for (int i = 0; i < 32; i++) { blkbase[i * NEXP + e] = b; b += blkcnt[i * NEXP + e]; }
  }
}

__global__ void __launch_bounds__(256) pos_kernel(const int* __restrict__ tk_idx,
                                                  const int* __restrict__ blkbase,
                                                  int* __restrict__ slot_pos) {
  __shared__ int base[NEXP];
  __shared__ int wcnt[8][NEXP];
  int t = threadIdx.x, lane = t & 31, w = t >> 5;
  if (t < NEXP) base[t] = blkbase[blockIdx.x * NEXP + t];
  __syncthreads();
  for (int p = 0; p < 4; p++) {
    int s = blockIdx.x * 1024 + p * 256 + t;
    int e = tk_idx[s];
    unsigned long long m[NEXP];
    #pragma unroll
    for (int q = 0; q < NEXP; q++) m[q] = __ballot(e == q);
    if (lane == 0) {
      #pragma unroll
      for (int q = 0; q < NEXP; q++) wcnt[w][q] = __popcll(m[q]);
    }
    int rank = __popcll(m[e] & ((1ull << lane) - 1ull));
    __syncthreads();
    int pre = 0;
    for (int w2 = 0; w2 < 8; w2++) if (w2 < w) pre += wcnt[w2][e];
    slot_pos[s] = base[e] + pre + rank;
    int tot = 0;
    if (t < NEXP) for (int w2 = 0; w2 < 8; w2++) tot += wcnt[w2][t];
    __syncthreads();
    if (t < NEXP) base[t] += tot;
    __syncthreads();
  }
}

// ---- scatter x -> exp_x (bf16), one wave per slot ----
__global__ void __launch_bounds__(256) scatter_kernel(const float* __restrict__ x,
                                                      const int* __restrict__ tk_idx,
                                                      const int* __restrict__ slot_pos,
                                                      unsigned short* __restrict__ expx) {
  int t = threadIdx.x, lane = t & 31, w = t >> 5;
  int s = blockIdx.x * 8 + w;
  int p = slot_pos[s];
  if (p >= CAP) return;
  int e = tk_idx[s];
  int n = s >> 1;
  const float4* src = (const float4*)(x + (size_t)n * D_MODEL);
  uint2* dst = (uint2*)(expx + ((size_t)e * CAP + p) * D_MODEL);
  #pragma unroll
  for (int i = 0; i < 8; i++) {
    float4 v = src[i * 32 + lane];
    uint2 o;
    o.x = (unsigned)f2bf(v.x) | ((unsigned)f2bf(v.y) << 16);
    o.y = (unsigned)f2bf(v.z) | ((unsigned)f2bf(v.w) << 16);
    dst[i * 32 + lane] = o;
  }
}

// ---- GEMM1 + SwiGLU: h = silu(X@Wa) * (X@Wg); async double-buffered LDS ----
__global__ void __launch_bounds__(256) gemm1_swiglu(const unsigned short* __restrict__ expx,
                                                    const unsigned short* __restrict__ bfc,
                                                    unsigned short* __restrict__ h) {
  __shared__ __align__(16) unsigned short sA[2][128 * LSTR];
  __shared__ __align__(16) unsigned short sBa[2][64 * LSTR];
  __shared__ __align__(16) unsigned short sBg[2][64 * LSTR];
  int e = blockIdx.z;
  int jA = blockIdx.x * 64;
  int M0 = blockIdx.y * 128;
  const unsigned short* A = expx + (size_t)e * CAP * D_MODEL;          // [c][k]
  const unsigned short* B = bfc + (size_t)e * D_FF * D_MODEL;          // [n][k]
  unsigned short* H = h + (size_t)e * CAP * D_HALF;
  int t = threadIdx.x, wid = t >> 5, lane = t & 31;
  int waveM = (wid >> 1) * 32, waveN = (wid & 1) * 32;
  int kp = (t & 7) * 8, rr = t >> 3;
  v8f accA[2][2] = {}; v8f accG[2][2] = {};

  auto stage = [&](int buf, int k0) {
    #pragma unroll
    for (int i = 0; i < 4; i++) {
      int r = rr + i * 32;
      async_b128(A, (unsigned)((M0 + r) * D_MODEL + k0 + kp), &sA[buf][r * LSTR + kp]);
    }
    #pragma unroll
    for (int i = 0; i < 2; i++) {
      int nn = rr + i * 32;
      async_b128(B, (unsigned)((jA + nn) * D_MODEL + k0 + kp), &sBa[buf][nn * LSTR + kp]);
      async_b128(B, (unsigned)((jA + D_HALF + nn) * D_MODEL + k0 + kp), &sBg[buf][nn * LSTR + kp]);
    }
  };

  stage(0, 0);
  const int NK = D_MODEL / 64;   // 16
  for (int it = 0; it < NK; ++it) {
    async_wait0();
    __syncthreads();             // buf[it&1] ready; buf[(it+1)&1] free to restage
    if (it + 1 < NK) stage((it + 1) & 1, (it + 1) * 64);
    const unsigned short* cA = sA[it & 1];
    const unsigned short* cBa = sBa[it & 1];
    const unsigned short* cBg = sBg[it & 1];
    #pragma unroll
    for (int kk = 0; kk < 64; kk += 32) {
      v16bf af[2], ba[2], bg[2];
      #pragma unroll
      for (int mi = 0; mi < 2; mi++) af[mi] = load_afrag(cA, waveM + mi * 16, kk);
      #pragma unroll
      for (int ni = 0; ni < 2; ni++) {
        ba[ni] = load_bfrag(cBa, waveN + ni * 16, kk);
        bg[ni] = load_bfrag(cBg, waveN + ni * 16, kk);
      }
      #pragma unroll
      for (int mi = 0; mi < 2; mi++)
        #pragma unroll
        for (int ni = 0; ni < 2; ni++) {
          accA[mi][ni] = wmma_bf16(af[mi], ba[ni], accA[mi][ni]);
          accG[mi][ni] = wmma_bf16(af[mi], bg[ni], accG[mi][ni]);
        }
    }
  }
  int rOff = (lane >> 4) << 3, cOff = lane & 15;
  #pragma unroll
  for (int mi = 0; mi < 2; mi++)
    #pragma unroll
    for (int ni = 0; ni < 2; ni++)
      #pragma unroll
      for (int i = 0; i < 8; i++) {
        int row = M0 + waveM + mi * 16 + i + rOff;
        int col = jA + waveN + ni * 16 + cOff;
        float a = accA[mi][ni][i], g = accG[mi][ni][i];
        float sv = a / (1.0f + __expf(-a));      // silu(a)
        H[(size_t)row * D_HALF + col] = f2bf(sv * g);
      }
}

// ---- GEMM2: exp_out = h @ c_proj; async double-buffered LDS, f32 out ----
__global__ void __launch_bounds__(256) gemm2_kernel(const unsigned short* __restrict__ h,
                                                    const unsigned short* __restrict__ bpr,
                                                    float* __restrict__ expo) {
  __shared__ __align__(16) unsigned short sA[2][128 * LSTR];
  __shared__ __align__(16) unsigned short sB[2][128 * LSTR];
  int e = blockIdx.z;
  int N0 = blockIdx.x * 128, M0 = blockIdx.y * 128;
  const unsigned short* A = h + (size_t)e * CAP * D_HALF;              // [c][f]
  const unsigned short* B = bpr + (size_t)e * D_MODEL * D_HALF;        // [n][f]
  float* O = expo + (size_t)e * CAP * D_MODEL;
  int t = threadIdx.x, wid = t >> 5, lane = t & 31;
  int waveM = (wid >> 1) * 32, waveN = (wid & 1) * 64;
  int kp = (t & 7) * 8, rr = t >> 3;
  v8f acc[2][4] = {};

  auto stage = [&](int buf, int k0) {
    #pragma unroll
    for (int i = 0; i < 4; i++) {
      int r = rr + i * 32;
      async_b128(A, (unsigned)((M0 + r) * D_HALF + k0 + kp), &sA[buf][r * LSTR + kp]);
      async_b128(B, (unsigned)((N0 + r) * D_HALF + k0 + kp), &sB[buf][r * LSTR + kp]);
    }
  };

  stage(0, 0);
  const int NK = D_HALF / 64;    // 24
  for (int it = 0; it < NK; ++it) {
    async_wait0();
    __syncthreads();
    if (it + 1 < NK) stage((it + 1) & 1, (it + 1) * 64);
    const unsigned short* cA = sA[it & 1];
    const unsigned short* cB = sB[it & 1];
    #pragma unroll
    for (int kk = 0; kk < 64; kk += 32) {
      v16bf af[2], bf[4];
      #pragma unroll
      for (int mi = 0; mi < 2; mi++) af[mi] = load_afrag(cA, waveM + mi * 16, kk);
      #pragma unroll
      for (int ni = 0; ni < 4; ni++) bf[ni] = load_bfrag(cB, waveN + ni * 16, kk);
      #pragma unroll
      for (int mi = 0; mi < 2; mi++)
        #pragma unroll
        for (int ni = 0; ni < 4; ni++)
          acc[mi][ni] = wmma_bf16(af[mi], bf[ni], acc[mi][ni]);
    }
  }
  int rOff = (lane >> 4) << 3, cOff = lane & 15;
  #pragma unroll
  for (int mi = 0; mi < 2; mi++)
    #pragma unroll
    for (int ni = 0; ni < 4; ni++)
      #pragma unroll
      for (int i = 0; i < 8; i++) {
        int row = M0 + waveM + mi * 16 + i + rOff;
        int col = N0 + waveN + ni * 16 + cOff;
        O[(size_t)row * D_MODEL + col] = acc[mi][ni][i];
      }
}

// ---- combine: out[n] = sum_k w[n,k] * exp_out[e_k, pos_k], one wave per token ----
__global__ void __launch_bounds__(256) combine_kernel(const float* __restrict__ expo,
                                                      const int* __restrict__ tk_idx,
                                                      const float* __restrict__ tk_w,
                                                      const int* __restrict__ slot_pos,
                                                      float* __restrict__ out) {
  int t = threadIdx.x, lane = t & 31, w = t >> 5;
  int n = blockIdx.x * 8 + w;
  float4 acc[8];
  #pragma unroll
  for (int i = 0; i < 8; i++) acc[i] = make_float4(0.f, 0.f, 0.f, 0.f);
  #pragma unroll
  for (int k = 0; k < TOPK; k++) {
    int s = n * TOPK + k;
    int p = slot_pos[s];
    if (p < CAP) {
      int e = tk_idx[s];
      float wv = tk_w[s];
      const float4* src = (const float4*)(expo + ((size_t)e * CAP + p) * D_MODEL);
      #pragma unroll
      for (int i = 0; i < 8; i++) {
        float4 v = src[i * 32 + lane];
        acc[i].x += wv * v.x; acc[i].y += wv * v.y;
        acc[i].z += wv * v.z; acc[i].w += wv * v.w;
      }
    }
  }
  float4* op = (float4*)(out + (size_t)n * D_MODEL);
  #pragma unroll
  for (int i = 0; i < 8; i++) op[i * 32 + lane] = acc[i];
}

extern "C" void kernel_launch(void* const* d_in, const int* in_sizes, int n_in,
                              void* d_out, int out_size, void* d_ws, size_t ws_size,
                              hipStream_t stream) {
  (void)in_sizes; (void)n_in; (void)out_size; (void)ws_size;
  const float* x   = (const float*)d_in[0];
  const float* wg  = (const float*)d_in[1];
  const float* cfc = (const float*)d_in[2];
  const float* cpr = (const float*)d_in[3];
  float* out = (float*)d_out;

  char* ws = (char*)d_ws;
  size_t off = 0;
  auto alloc = [&](size_t bytes) -> char* {
    char* p = ws + off;
    off += (bytes + 255) & ~(size_t)255;
    return p;
  };
  unsigned short* cfcT = (unsigned short*)alloc((size_t)NEXP * D_FF * D_MODEL * 2);
  unsigned short* cprT = (unsigned short*)alloc((size_t)NEXP * D_MODEL * D_HALF * 2);
  unsigned short* expx = (unsigned short*)alloc((size_t)NEXP * CAP * D_MODEL * 2);
  unsigned short* hbuf = (unsigned short*)alloc((size_t)NEXP * CAP * D_HALF * 2);
  float* expo          = (float*)alloc((size_t)NEXP * CAP * D_MODEL * 4);
  int*   tk_idx        = (int*)alloc((size_t)NSLOT * 4);
  float* tk_w          = (float*)alloc((size_t)NSLOT * 4);
  int*   slot_pos      = (int*)alloc((size_t)NSLOT * 4);
  int*   blkcnt        = (int*)alloc(32 * NEXP * 4);
  int*   blkbase       = (int*)alloc(32 * NEXP * 4);

  transpose_convert<<<dim3(D_FF / 32, D_MODEL / 32, NEXP), dim3(32, 8), 0, stream>>>(
      cfc, cfcT, D_MODEL, D_FF);
  transpose_convert<<<dim3(D_MODEL / 32, D_HALF / 32, NEXP), dim3(32, 8), 0, stream>>>(
      cpr, cprT, D_HALF, D_MODEL);
  gate_kernel<<<N_TOK / 8, 256, 0, stream>>>(x, wg, tk_idx, tk_w);
  hist_kernel<<<32, 256, 0, stream>>>(tk_idx, blkcnt);
  scan_kernel<<<1, 32, 0, stream>>>(blkcnt, blkbase);
  pos_kernel<<<32, 256, 0, stream>>>(tk_idx, blkbase, slot_pos);
  scatter_kernel<<<NSLOT / 8, 256, 0, stream>>>(x, tk_idx, slot_pos, expx);
  gemm1_swiglu<<<dim3(D_HALF / 64, CAP / 128, NEXP), 256, 0, stream>>>(expx, cfcT, hbuf);
  gemm2_kernel<<<dim3(D_MODEL / 128, CAP / 128, NEXP), 256, 0, stream>>>(hbuf, cprT, expo);
  combine_kernel<<<N_TOK / 8, 256, 0, stream>>>(expo, tk_idx, tk_w, slot_pos, out);
}
